// TimeAttention_69595650064559
// MI455X (gfx1250) — matrix operand order repaired and checked by reading, not compiled, not run on hardware
//
#include <hip/hip_runtime.h>
#include <hip/hip_bf16.h>

typedef __attribute__((ext_vector_type(16))) __bf16 v16bf;
typedef __attribute__((ext_vector_type(8)))  float  v8f;
typedef __attribute__((ext_vector_type(4)))  unsigned int u32x4;
typedef __attribute__((ext_vector_type(8)))  int  i32x8;
typedef __attribute__((ext_vector_type(4)))  int  i32x4;

#define S_LEN   2048
#define DHEAD   64
#define NHEADS  16
#define NEG_INF_F (-1e9f)

// native RNE convert -> v_cvt_pk_bf16_f32
__device__ __forceinline__ __bf16 f2bf(float f) { return (__bf16)f; }

// A-fragment (16x32 bf16): lane 0-15 -> row M=lane, K {2*c0..2*c0+15} (contiguous),
// c0=0; lanes 16-31 -> row M=lane-16, c0=8.  k0 selects the K chunk (0 or 32).
__device__ __forceinline__ v16bf make_afrag(const float* rowp, int c0, int k0) {
  const float4* p0 = (const float4*)(rowp + k0 + 2 * c0);
  const float4* p1 = (const float4*)(rowp + k0 + 2 * c0 + 8);
  float4 a = p0[0], b = p0[1], c = p1[0], d = p1[1];
  v16bf r;
  r[0]=f2bf(a.x); r[1]=f2bf(a.y); r[2]=f2bf(a.z); r[3]=f2bf(a.w);
  r[4]=f2bf(b.x); r[5]=f2bf(b.y); r[6]=f2bf(b.z); r[7]=f2bf(b.w);
  r[8]=f2bf(c.x); r[9]=f2bf(c.y); r[10]=f2bf(c.z); r[11]=f2bf(c.w);
  r[12]=f2bf(d.x); r[13]=f2bf(d.y); r[14]=f2bf(d.z); r[15]=f2bf(d.w);
  return r;
}

// B-fragment (32x16 bf16): lane n (n=lane&15, h16 = 0|16 by lane half) holds
// column N=n, K = k0+h16 .. k0+h16+15 (contiguous).  B = (K+T)^T -> fused add.
__device__ __forceinline__ v16bf make_bfrag_kt(const float* krow, const float* trow,
                                               int h16, int k0) {
  const float4* kp = (const float4*)(krow + k0 + h16);
  const float4* tp = (const float4*)(trow + k0 + h16);
  v16bf r;
  #pragma unroll
  for (int j = 0; j < 4; ++j) {
    float4 kv = kp[j], tv = tp[j];
    r[4*j+0] = f2bf(kv.x + tv.x);
    r[4*j+1] = f2bf(kv.y + tv.y);
    r[4*j+2] = f2bf(kv.z + tv.z);
    r[4*j+3] = f2bf(kv.w + tv.w);
  }
  return r;
}

// ---- Tensor Data Mover: load one 64x64 fp32 tile (16 KB) into LDS ----------
// D# built per CDNA5 ISA ch.8: group0 = {count|flags, lds_addr, global_addr,
// addr_hi|type=2}; group1 packs data_size=4B, tensor 64x64, tile 64x64,
// dim0 stride 64.  Groups 2/3 unused (2D tensor).
__device__ __forceinline__ void tdm_load_v_tile(const float* gsrc, unsigned lds_off) {
  unsigned long long ga = (unsigned long long)(size_t)gsrc;
  u32x4 g0;
  g0[0] = 1u;                                   // count=1, no gather, user desc
  g0[1] = lds_off;                              // lds_addr (bytes)
  g0[2] = (unsigned)(ga & 0xFFFFFFFFu);         // global_addr[31:0]
  g0[3] = (unsigned)((ga >> 32) & 0x01FFFFFFu)  // global_addr[56:32]
        | (2u << 30);                           // type=2 ("image")
  i32x8 g1;
  g1[0] = (int)(2u << 16);                      // data_size = 4 bytes
  g1[1] = (int)(64u << 16);                     // tensor_dim0 = 64 (bits 79:48 lo)
  g1[2] = (int)(64u << 16);                     // tensor_dim1 = 64 (bits 111:80 lo)
  g1[3] = (int)(64u << 16);                     // tile_dim0 = 64 (bits 127:112)
  g1[4] = 64;                                   // tile_dim1 = 64, tile_dim2 = 0
  g1[5] = 64;                                   // tensor_dim0_stride = 64
  g1[6] = 0;                                    // stride hi / dim1_stride lo
  g1[7] = 0;
  i32x4 g2 = {0, 0, 0, 0};
  i32x4 g3 = {0, 0, 0, 0};
#if defined(__clang_major__) && (__clang_major__ >= 23)
  i32x8 g4 = {0, 0, 0, 0, 0, 0, 0, 0};
  __builtin_amdgcn_tensor_load_to_lds(g0, g1, g2, g3, g4, 0);
#else
  __builtin_amdgcn_tensor_load_to_lds(g0, g1, g2, g3, 0);
#endif
}

__global__ __launch_bounds__(128)
void time_attn_kernel(const float* __restrict__ Q, const float* __restrict__ K,
                      const float* __restrict__ V, const float* __restrict__ T,
                      const int* __restrict__ mask,
                      float* __restrict__ out, float* __restrict__ pattn) {
  extern __shared__ char smem[];
  float*  sS = (float*)smem;                                 // 16 x 2048 fp32 (128 KB)
  __bf16* sP = (__bf16*)(smem + 16 * S_LEN * sizeof(float)); // 16 x 2048 bf16 ( 64 KB)
  float*  sV = (float*)(smem + 16 * S_LEN * 6);              // 2 x 64 x 64 fp32 (32 KB)

  const int tid  = threadIdx.x;
  const int lane = tid & 31;
  const int wave = tid >> 5;                 // 0..3
  const int qt   = blockIdx.x & (S_LEN / 16 - 1);
  const int bh   = blockIdx.x >> 7;          // / (S_LEN/16)
  const int b    = bh / NHEADS;
  const int q0   = qt * 16;

  const float* Qb = Q + ((size_t)bh * S_LEN + q0) * DHEAD;
  const float* Kb = K + (size_t)bh * S_LEN * DHEAD;
  const float* Tb = T + (size_t)bh * S_LEN * DHEAD;
  const float* Vb = V + (size_t)bh * S_LEN * DHEAD;
  const int*   Mb = mask + ((size_t)b * S_LEN + q0) * S_LEN;
  float* Ob = out   + ((size_t)bh * S_LEN + q0) * DHEAD;
  float* Pb = pattn + ((size_t)bh * S_LEN + q0) * S_LEN;

  const int mrow = lane & 15;                // M row (A) / N column (B)
  const int c0   = (lane < 16) ? 0 : 8;      // A-fragment K sub-offset
  const int h16  = (lane < 16) ? 0 : 16;     // B-fragment K half

  // ---- Q A-fragments (kept in VGPRs for all of phase 1) ----
  const float* qrow = Qb + mrow * DHEAD;
  v16bf qf0 = make_afrag(qrow, c0, 0);
  v16bf qf1 = make_afrag(qrow, c0, 32);

  // ---- Phase 1: scores = Q (K+T)^T / 8  -> LDS ----
  for (int kt = wave; kt < S_LEN / 16; kt += 4) {
    const int kbase = kt * 16;
    const float* krow = Kb + (size_t)(kbase + mrow) * DHEAD;
    const float* trow = Tb + (size_t)(kbase + mrow) * DHEAD;
    if (kt + 4 < S_LEN / 16) {               // CDNA5 global_prefetch_b8
      __builtin_prefetch(krow + 64 * DHEAD, 0, 3);
      __builtin_prefetch(trow + 64 * DHEAD, 0, 3);
    }
    v16bf bf0 = make_bfrag_kt(krow, trow, h16, 0);
    v16bf bf1 = make_bfrag_kt(krow, trow, h16, 32);
    v8f c = {};
    c = __builtin_amdgcn_wmma_f32_16x16x32_bf16(false, qf0, false, bf0, (short)0, c, false, false);
    c = __builtin_amdgcn_wmma_f32_16x16x32_bf16(false, qf1, false, bf1, (short)0, c, false, false);
    const int n     = lane & 15;
    const int mbase = (lane < 16) ? 0 : 8;   // C layout: VGPR i -> M=i / M=8+i
    #pragma unroll
    for (int i = 0; i < 8; ++i)
      sS[(mbase + i) * S_LEN + kbase + n] = c[i] * 0.125f;
  }
  __syncthreads();

  // ---- Phase 2: masked softmax (fp32), write p_attn (NT), bf16 P -> LDS ----
  for (int r = wave * 4; r < wave * 4 + 4; ++r) {
    float* srow = sS + r * S_LEN;
    const int* mr = Mb + (size_t)r * S_LEN;
    float mx = -3.4e38f;
    for (int k = lane; k < S_LEN; k += 32) {
      int mv = __builtin_nontemporal_load(mr + k);        // read-once -> TH_NT
      float s = (mv == 0) ? NEG_INF_F : srow[k];
      srow[k] = s;
      mx = fmaxf(mx, s);
    }
    #pragma unroll
    for (int o = 16; o > 0; o >>= 1) mx = fmaxf(mx, __shfl_xor(mx, o, 32));
    float sum = 0.f;
    for (int k = lane; k < S_LEN; k += 32) {
      float p = __expf(srow[k] - mx);
      srow[k] = p;
      sum += p;
    }
    #pragma unroll
    for (int o = 16; o > 0; o >>= 1) sum += __shfl_xor(sum, o, 32);
    const float inv = 1.0f / sum;
    __bf16* prow = sP + r * S_LEN;
    float*  gp   = Pb + (size_t)r * S_LEN;
    for (int k = lane; k < S_LEN; k += 32) {
      float p = srow[k] * inv;
      __builtin_nontemporal_store(p, gp + k);             // 536 MB write-once
      prow[k] = f2bf(p);                                  // stream: keep out of L2
    }
  }
  __syncthreads();

  // ---- Phase 3: out = P V with TDM double-buffered V staging ----
  // LDS byte offsets of the two staging buffers (flat addr low 32 bits == LDS offset)
  const unsigned vs_off0 = (unsigned)(size_t)(void*)sV;
  const unsigned vs_off1 = vs_off0 + 64 * DHEAD * sizeof(float);
  const int col0 = wave * 16;
  const int n    = lane & 15;
  v8f acc = {};

  if (wave == 0) tdm_load_v_tile(Vb, vs_off0);          // prologue: tile 0
  for (int t = 0; t < S_LEN / 64; ++t) {
    if (wave == 0) {
      if (t + 1 < S_LEN / 64) {                         // issue next, wait current
        tdm_load_v_tile(Vb + (size_t)(t + 1) * 64 * DHEAD,
                        ((t + 1) & 1) ? vs_off1 : vs_off0);
        __builtin_amdgcn_s_wait_tensorcnt(1);
      } else {
        __builtin_amdgcn_s_wait_tensorcnt(0);
      }
    }
    __syncthreads();                                    // tile t visible to all waves
    const float* Vt = sV + (size_t)(t & 1) * 64 * DHEAD;
    #pragma unroll
    for (int cch = 0; cch < 2; ++cch) {
      const int kk = t * 64 + cch * 32;
      const __bf16* prow = sP + mrow * S_LEN + kk;      // A-frag from LDS probs
      v16bf af;
      #pragma unroll
      for (int j = 0; j < 8; ++j) af[j]     = prow[2 * c0 + j];
      #pragma unroll
      for (int j = 0; j < 8; ++j) af[8 + j] = prow[2 * c0 + 8 + j];
      v16bf bf;                                         // B-frag from staged V
      const float* vcol = Vt + (size_t)(cch * 32 + h16) * DHEAD + col0 + n;
      #pragma unroll
      for (int j = 0; j < 16; ++j) bf[j] = f2bf(vcol[(size_t)j * DHEAD]);
      acc = __builtin_amdgcn_wmma_f32_16x16x32_bf16(false, af, false, bf, (short)0, acc, false, false);
    }
    __syncthreads();                                    // before buffer reuse
  }
  {
    const int mbase = (lane < 16) ? 0 : 8;
    #pragma unroll
    for (int i = 0; i < 8; ++i)
      __builtin_nontemporal_store(acc[i], Ob + (size_t)(mbase + i) * DHEAD + col0 + n);
  }
}

extern "C" void kernel_launch(void* const* d_in, const int* in_sizes, int n_in,
                              void* d_out, int out_size, void* d_ws, size_t ws_size,
                              hipStream_t stream) {
  (void)in_sizes; (void)n_in; (void)d_ws; (void)ws_size; (void)out_size;
  const float* Q    = (const float*)d_in[0];
  const float* K    = (const float*)d_in[1];
  const float* V    = (const float*)d_in[2];
  const float* T    = (const float*)d_in[3];
  const int*   mask = (const int*)d_in[4];

  float* out   = (float*)d_out;
  float* pattn = out + (size_t)2 * NHEADS * S_LEN * DHEAD;   // tuple order: (out, p_attn)

  const int nblocks = 2 * NHEADS * (S_LEN / 16);             // 4096
  const size_t lds  = (size_t)16 * S_LEN * sizeof(float)     // scores, 128 KB
                    + (size_t)16 * S_LEN * 2                 // bf16 P,   64 KB
                    + (size_t)2 * 64 * DHEAD * sizeof(float);// V stage,  32 KB
  time_attn_kernel<<<dim3(nblocks), dim3(128), lds, stream>>>(Q, K, V, T, mask, out, pattn);
}
